// PAM_10934986736343
// MI455X (gfx1250) — matrix operand order repaired and checked by reading, not compiled, not run on hardware
//
#include <hip/hip_runtime.h>

typedef __attribute__((ext_vector_type(16))) _Float16 v16h;
typedef __attribute__((ext_vector_type(8)))  _Float16 v8h;
typedef __attribute__((ext_vector_type(8)))  float    v8f;
typedef __attribute__((ext_vector_type(4)))  unsigned int u32x4;
typedef __attribute__((ext_vector_type(8)))  int          i32x8;
typedef __attribute__((ext_vector_type(4)))  int          i32x4;

union Frag16 { v16h v; v8h r[2]; _Float16 h[16]; };

#define B_    4
#define S_    2048
#define D_    256
#define H_    8
#define DK_   32
#define SCALE_ 0.42044820762685725f   // 1 / 32^0.25
#define NEG_BIG (-1.0e7f)
#define KPAD  40                      // f16 P-scratch row stride (80B, 16B-aligned runs)
#define KSTR  32                      // f32 K/V LDS row stride (128B rows)
#define NCH   (S_ / 32)

#if defined(__has_builtin)
#if __has_builtin(__builtin_amdgcn_tensor_load_to_lds)
#define HAVE_TDM 1
#endif
#endif
#ifndef HAVE_TDM
#define HAVE_TDM 0
#endif

#if HAVE_TDM
// Issue one 2-D TDM load: 32 rows x 32 f32, global row stride D_=256 floats,
// packed contiguously into LDS at lds_off. D# per CDNA5 ISA ch.8.
__device__ __forceinline__ void tdm_load_2d(unsigned int lds_off, const float* gptr) {
    unsigned long long ga = (unsigned long long)(uintptr_t)gptr;
    u32x4 g0 = {
        1u,                                   // count=1 (valid user descriptor)
        lds_off,                              // lds_addr [63:32]
        (unsigned int)ga,                     // global_addr low
        ((unsigned int)(ga >> 32) & 0x01FFFFFFu) | 0x80000000u  // addr hi | type=2
    };
    i32x8 g1 = {
        (int)(2u << 16),                      // data_size = 4B; no pad/iterate/multicast
        (int)(32u << 16),                     // tensor_dim0 = 32 (low16 at bits 63:48)
        (int)(32u << 16),                     // tensor_dim0 hi=0 | tensor_dim1=32 low16
        (int)(32u << 16),                     // tensor_dim1 hi=0 | tile_dim0 = 32
        32,                                   // tile_dim1 = 32, tile_dim2 = 0
        256,                                  // tensor_dim0_stride = D_ (low 32)
        0, 0
    };
    i32x4 z4 = {};
#if __clang_major__ >= 23
    i32x8 z8 = {};
    __builtin_amdgcn_tensor_load_to_lds(g0, g1, z4, z4, z8, 0);
#else
    __builtin_amdgcn_tensor_load_to_lds(g0, g1, z4, z4, 0);
#endif
}
#endif

// 16 contiguous f32 -> f16 fragment half
__device__ __forceinline__ void cvt16c(Frag16& f, const float* p) {
    float4 a = ((const float4*)p)[0];
    float4 b = ((const float4*)p)[1];
    float4 c = ((const float4*)p)[2];
    float4 d = ((const float4*)p)[3];
    float t[16] = {a.x,a.y,a.z,a.w, b.x,b.y,b.z,b.w, c.x,c.y,c.z,c.w, d.x,d.y,d.z,d.w};
#pragma unroll
    for (int i = 0; i < 16; ++i) f.h[i] = (_Float16)t[i];
}
// 16 strided f32 -> f16 (transposing read; lanes sweep consecutive columns -> no bank conflicts)
__device__ __forceinline__ void cvt16s(Frag16& f, const float* p, int stride) {
#pragma unroll
    for (int i = 0; i < 16; ++i) f.h[i] = (_Float16)p[i * stride];
}

// ---------------------------------------------------------------------------
// Kernel 1: flash attention (Q=K=heads(pos), V=heads(x)), f16 out to ws.
// grid = B*H*(S/64) blocks of 128 threads (4 waves, 16 queries per wave).
// K/V 32-key chunks DMA'd into double-buffered LDS by the TDM.
// ---------------------------------------------------------------------------
__global__ __launch_bounds__(128) void attn_kernel(
    const float* __restrict__ x, const float* __restrict__ mask,
    const float* __restrict__ pos, _Float16* __restrict__ attn_out)
{
    __shared__ __align__(16) float    Kf[2][32 * KSTR];     // K chunk [key][dk] f32
    __shared__ __align__(16) float    Vf[2][32 * KSTR];     // V chunk [key][d]  f32
    __shared__ __align__(16) _Float16 PS[4 * 16 * KPAD];    // per-wave P scratch [q][key]

    const int tid  = threadIdx.x;
    const int wid  = tid >> 5;
    const int lane = tid & 31;
    const int l16  = lane & 15;
    const int hh   = lane >> 4;          // half-wave: 0 or 1

    const int bid  = blockIdx.x;
    const int qblk = bid & 31;           // S/64 = 32
    const int h    = (bid >> 5) & 7;
    const int b    = bid >> 8;
    const int q0   = qblk * 64 + wid * 16;
    const int hd   = h * DK_;

    const float* kbase = pos + ((size_t)b * S_) * D_ + hd;
    const float* vbase = x   + ((size_t)b * S_) * D_ + hd;

    // ---- Q fragment: A-matrix 16x32 f16 (row = lane%16, K split by half-wave)
    Frag16 qa;
    {
        const float* qp = pos + ((size_t)(b * S_ + q0 + l16)) * D_ + hd;
        const int d0 = hh ? 16 : 0;
        float4 f0 = *(const float4*)(qp + d0);
        float4 f1 = *(const float4*)(qp + d0 + 4);
        float4 f2 = *(const float4*)(qp + d0 + 8);
        float4 f3 = *(const float4*)(qp + d0 + 12);
        float t0[8] = {f0.x, f0.y, f0.z, f0.w, f1.x, f1.y, f1.z, f1.w};
        float t1[8] = {f2.x, f2.y, f2.z, f2.w, f3.x, f3.y, f3.z, f3.w};
#pragma unroll
        for (int i = 0; i < 8; ++i) {
            qa.h[i]     = (_Float16)t0[i];
            qa.h[8 + i] = (_Float16)t1[i];
        }
    }

    // query-side mask bits for the 8 C-layout rows this lane holds
    bool mq[8];
#pragma unroll
    for (int e = 0; e < 8; ++e)
        mq[e] = mask[(size_t)b * S_ + q0 + e + 8 * hh] != 0.0f;

    float mrun[8], lrun[8];
#pragma unroll
    for (int e = 0; e < 8; ++e) { mrun[e] = -1.0e30f; lrun[e] = 0.0f; }
    v8f a0 = {}, a1 = {};

    _Float16* Pw = &PS[wid * 16 * KPAD];

#if HAVE_TDM
    const bool issuer = (__builtin_amdgcn_readfirstlane(wid) == 0);  // scalar branch
    const unsigned int ldsK[2] = { (unsigned int)(uintptr_t)&Kf[0][0],
                                   (unsigned int)(uintptr_t)&Kf[1][0] };
    const unsigned int ldsV[2] = { (unsigned int)(uintptr_t)&Vf[0][0],
                                   (unsigned int)(uintptr_t)&Vf[1][0] };
    if (issuer) {                      // prologue: chunk 0 into buffer 0
        tdm_load_2d(ldsK[0], kbase);
        tdm_load_2d(ldsV[0], vbase);
    }
#endif

    for (int ic = 0; ic < NCH; ++ic) {
        const int buf = ic & 1;
        const int kc  = ic * 32;
#if HAVE_TDM
        if (issuer) __builtin_amdgcn_s_wait_tensorcnt(0);   // chunk ic landed
        __syncthreads();                                    // publish; prior reads of buf^1 done
        if (issuer && ic + 1 < NCH) {                       // pipeline chunk ic+1
            tdm_load_2d(ldsK[buf ^ 1], kbase + (size_t)(kc + 32) * D_);
            tdm_load_2d(ldsV[buf ^ 1], vbase + (size_t)(kc + 32) * D_);
        }
#else
        __syncthreads();
        {   // fallback: cooperative f32 staging, 8 floats per thread per array
            const int li = tid * 8;
            const int kk = li >> 5, d = li & 31;
            const float* ks = kbase + (size_t)(kc + kk) * D_ + d;
            const float* vs = vbase + (size_t)(kc + kk) * D_ + d;
            float4* kd = (float4*)&Kf[buf][kk * KSTR + d];
            kd[0] = ((const float4*)ks)[0]; kd[1] = ((const float4*)ks)[1];
            float4* vd = (float4*)&Vf[buf][kk * KSTR + d];
            vd[0] = ((const float4*)vs)[0]; vd[1] = ((const float4*)vs)[1];
        }
        __syncthreads();
#endif
        const float* Kb = &Kf[buf][0];
        const float* Vb = &Vf[buf][0];

        // B fragments for the two 16-key score tiles (col = lane%16, K-dim by half-wave)
        Frag16 kb0, kb1;
        cvt16c(kb0, Kb + l16 * KSTR + hh * 16);
        cvt16c(kb1, Kb + (16 + l16) * KSTR + hh * 16);

        v8f z = {};
        v8f c0 = __builtin_amdgcn_wmma_f32_16x16x32_f16(false, qa.v, false, kb0.v, (short)0, z, false, false);
        v8f c1 = __builtin_amdgcn_wmma_f32_16x16x32_f16(false, qa.v, false, kb1.v, (short)0, z, false, false);

        const bool mk0 = mask[(size_t)b * S_ + kc + l16] != 0.0f;
        const bool mk1 = mask[(size_t)b * S_ + kc + 16 + l16] != 0.0f;

        float s0[8], s1[8], tmax[8];
#pragma unroll
        for (int e = 0; e < 8; ++e) {
            float u0 = c0[e] * SCALE_; if (!(mq[e] && mk0)) u0 = NEG_BIG;
            float u1 = c1[e] * SCALE_; if (!(mq[e] && mk1)) u1 = NEG_BIG;
            s0[e] = u0; s1[e] = u1;
            tmax[e] = fmaxf(u0, u1);
        }
#pragma unroll
        for (int off = 1; off < 16; off <<= 1)
#pragma unroll
            for (int e = 0; e < 8; ++e)
                tmax[e] = fmaxf(tmax[e], __shfl_xor(tmax[e], off, 32));

        float rs[8];
#pragma unroll
        for (int e = 0; e < 8; ++e) {
            float nm = fmaxf(mrun[e], tmax[e]);
            float f  = __expf(mrun[e] - nm);
            float p0 = __expf(s0[e] - nm);
            float p1 = __expf(s1[e] - nm);
            mrun[e] = nm;
            a0[e] *= f; a1[e] *= f;
            lrun[e] *= f;
            rs[e] = p0 + p1;
            const int r = e + 8 * hh;                 // C-layout row
            Pw[r * KPAD + l16]      = (_Float16)p0;
            Pw[r * KPAD + 16 + l16] = (_Float16)p1;
        }
#pragma unroll
        for (int off = 1; off < 16; off <<= 1)
#pragma unroll
            for (int e = 0; e < 8; ++e)
                rs[e] += __shfl_xor(rs[e], off, 32);
#pragma unroll
        for (int e = 0; e < 8; ++e) lrun[e] += rs[e];

        // P back as A fragment (same-wave LDS ops are in-order)
        Frag16 pa;
        {
            const int d0 = hh ? 16 : 0;
            pa.r[0] = *(const v8h*)&Pw[l16 * KPAD + d0];
            pa.r[1] = *(const v8h*)&Pw[l16 * KPAD + d0 + 8];
        }
        // V B fragments: column d fixed per lane, k sweeps (transposing strided read)
        Frag16 vb0, vb1;
        cvt16s(vb0, Vb + (hh * 16) * KSTR + l16,      KSTR);
        cvt16s(vb1, Vb + (hh * 16) * KSTR + 16 + l16, KSTR);

        a0 = __builtin_amdgcn_wmma_f32_16x16x32_f16(false, pa.v, false, vb0.v, (short)0, a0, false, false);
        a1 = __builtin_amdgcn_wmma_f32_16x16x32_f16(false, pa.v, false, vb1.v, (short)0, a1, false, false);
    }

#pragma unroll
    for (int e = 0; e < 8; ++e) {
        const int r = e + 8 * hh;
        const float inv = 1.0f / lrun[e];
        const size_t base = ((size_t)(b * S_ + q0 + r)) * D_ + hd;
        attn_out[base + l16]      = (_Float16)(a0[e] * inv);
        attn_out[base + 16 + l16] = (_Float16)(a1[e] * inv);
    }
}

// ---------------------------------------------------------------------------
// Kernel 2: relu(attn_out @ W + b), 8192x256x256, f16 WMMA, fp32 out
// grid = (8192/64, 256/16) blocks of 128 threads; W^T tile staged in LDS
// ---------------------------------------------------------------------------
#define WPAD 264
__global__ __launch_bounds__(128) void proj_kernel(
    const _Float16* __restrict__ attn, const float* __restrict__ W,
    const float* __restrict__ bias, float* __restrict__ out)
{
    __shared__ __align__(16) _Float16 WT[16 * WPAD];   // W^T tile [j][k]
    __shared__ float bs[16];

    const int tid  = threadIdx.x;
    const int wid  = tid >> 5;
    const int lane = tid & 31;
    const int l16  = lane & 15;
    const int hh   = lane >> 4;
    const int j0   = blockIdx.y * 16;
    const int row0 = blockIdx.x * 64 + wid * 16;

#pragma unroll
    for (int rep = 0; rep < 32; ++rep) {
        const int e = rep * 128 + tid;
        const int j = e & 15, k = e >> 4;
        WT[j * WPAD + k] = (_Float16)W[k * D_ + j0 + j];
    }
    if (tid < 16) bs[tid] = bias[j0 + tid];
    __syncthreads();

    v8f c = {};
    const _Float16* arow = attn + (size_t)(row0 + l16) * D_;
#pragma unroll
    for (int kk = 0; kk < 8; ++kk) {
        const int base = kk * 32;
        const int d0 = hh ? 8 : 0;
        Frag16 a, bf;
        a.r[0]  = *(const v8h*)(arow + base + d0);
        a.r[1]  = *(const v8h*)(arow + base + d0 + 16);
        bf.r[0] = *(const v8h*)&WT[l16 * WPAD + base + hh * 8];
        bf.r[1] = *(const v8h*)&WT[l16 * WPAD + base + hh * 8 + 16];
        c = __builtin_amdgcn_wmma_f32_16x16x32_f16(false, a.v, false, bf.v, (short)0, c, false, false);
    }

    const float bj = bs[l16];
#pragma unroll
    for (int e = 0; e < 8; ++e) {
        const int r = e + 8 * hh;
        float v = c[e] + bj;
        out[(size_t)(row0 + r) * D_ + j0 + l16] = v > 0.0f ? v : 0.0f;
    }
}

// ---------------------------------------------------------------------------
extern "C" void kernel_launch(void* const* d_in, const int* in_sizes, int n_in,
                              void* d_out, int out_size, void* d_ws, size_t ws_size,
                              hipStream_t stream) {
    (void)in_sizes; (void)n_in; (void)out_size; (void)ws_size;
    const float* x    = (const float*)d_in[0];
    const float* mask = (const float*)d_in[1];
    const float* pos  = (const float*)d_in[2];
    const float* W    = (const float*)d_in[3];
    const float* bias = (const float*)d_in[4];
    _Float16* attn = (_Float16*)d_ws;      // 8192*256 f16 = 4 MB scratch
    float* out = (float*)d_out;

    attn_kernel<<<dim3(B_ * H_ * (S_ / 64)), dim3(128), 0, stream>>>(x, mask, pos, attn);
    proj_kernel<<<dim3((B_ * S_) / 64, D_ / 16), dim3(128), 0, stream>>>(attn, W, bias, out);
}
